// Attention_35957466202456
// MI455X (gfx1250) — compile-verified, matrix operand order checked by reading, not run on hardware
//
#include <hip/hip_runtime.h>
#include <hip/hip_bf16.h>
#include <stdint.h>

// B=16, T=2048, D=512, HD=64 causal single-head attention, fp32 in/out.
// Phase 1: fused QKV projection. One block = 16 rows of x; Wq/Wk/Wv K-chunks
//          are async-copied (global_load_async_to_lds_b128) into LDS, x is
//          loaded once and reused for 3 WMMAs per K-step. bf16 q/k/v -> ws.
// Phase 2: flash attention, 1 wave per 16-query tile, 32-key chunks. K and V
//          tiles async-copied to LDS; S = QK^T and O += P V via
//          v_wmma_f32_16x16x32_bf16 with online softmax in between.

typedef __attribute__((ext_vector_type(16))) __bf16 v16bf;
typedef __attribute__((ext_vector_type(8)))  __bf16 v8bf;
typedef __attribute__((ext_vector_type(8)))  float  v8f;

#define BB  16
#define TT  2048
#define DDM 512
#define HDD 64

// Low 32 bits of a flat shared pointer == LDS byte address (aperture mapping).
__device__ __forceinline__ unsigned lds_addr(const void* p) {
  return (unsigned)(uintptr_t)p;
}

// Issue a per-lane 16B async copy global -> LDS (tracked by ASYNCcnt).
__device__ __forceinline__ void async_copy_b128(unsigned lds, uint64_t gaddr) {
  asm volatile("global_load_async_to_lds_b128 %0, %1, off"
               :: "v"(lds), "v"(gaddr) : "memory");
}
__device__ __forceinline__ void wait_async0() {
  asm volatile("s_wait_asynccnt 0x0" ::: "memory");
}

// A/B fragment for wmma_f32_16x16x32_bf16 (CDNA5 ISA 7.12.2):
// lane (m = lane&15, h = lane>>4) holds K = {8h..8h+7, 16+8h..16+8h+7} of
// row m (A) / column m (B). p points at element (m, 8h) of a row-major
// (contraction-contiguous) tile => two 16B loads.
__device__ __forceinline__ v16bf load_frag16(const __bf16* p) {
  union { v16bf v; v8bf h2[2]; } u;
  u.h2[0] = *(const v8bf*)p;
  u.h2[1] = *(const v8bf*)(p + 16);
  return u.v;
}

// ---------------- Phase 1: {q,k,v} = x @ W{q,k,v}  (M=32768, K=512, N=64) --
// Block = 128 threads (4 waves); wave w owns output columns 16w..16w+15.
__global__ __launch_bounds__(128) void qkv_proj_kernel(
    const float* __restrict__ x, const float* __restrict__ Wq,
    const float* __restrict__ Wk, const float* __restrict__ Wv,
    __bf16* __restrict__ qkv) {
  __shared__ alignas(16) float wS[3 * 32 * HDD];  // 3 x 8KB W K-chunks
  const int M = BB * TT;
  const int tile = blockIdx.x;          // 0..2047, 16 rows each
  const int t    = threadIdx.x;
  const int wave = t >> 5;
  const int lane = t & 31;
  const int row  = tile * 16;
  const int colB = wave * 16;
  const int ln   = lane & 15;
  const int h    = lane >> 4;
  const float* Ws[3] = {Wq, Wk, Wv};

  const v8f zero = {0.f, 0.f, 0.f, 0.f, 0.f, 0.f, 0.f, 0.f};
  v8f acc[3] = {zero, zero, zero};
  const unsigned ldsW = lds_addr(wS);

  for (int kk = 0; kk < DDM; kk += 32) {
    // Async-fill 3 contiguous 8KB chunks: W[kk..kk+31][0..63] (fp32).
#pragma unroll
    for (int w = 0; w < 3; ++w) {
      const uint64_t ga = (uint64_t)(uintptr_t)(Ws[w] + (size_t)kk * HDD);
      const unsigned lb = ldsW + (unsigned)w * 8192u;
#pragma unroll
      for (int j = 0; j < 4; ++j) {
        const unsigned off = (unsigned)(j * 128 + t) * 16u;
        async_copy_b128(lb + off, ga + off);
      }
    }
    wait_async0();
    __syncthreads();

    // A fragment straight from global x (fp32 -> bf16), reused 3x.
    const float* px = x + (size_t)(row + ln) * DDM + kk + 8 * h;
    v16bf a;
#pragma unroll
    for (int i = 0; i < 8; ++i) {
      a[i]     = (__bf16)px[i];
      a[8 + i] = (__bf16)px[16 + i];
    }

#pragma unroll
    for (int w = 0; w < 3; ++w) {
      const float* wp = wS + w * 32 * HDD;  // rows = K (0..31), cols = hd
      v16bf bf;
#pragma unroll
      for (int i = 0; i < 8; ++i) {
        bf[i]     = (__bf16)wp[(8 * h + i) * HDD + colB + ln];
        bf[8 + i] = (__bf16)wp[(16 + 8 * h + i) * HDD + colB + ln];
      }
      acc[w] = __builtin_amdgcn_wmma_f32_16x16x32_bf16(
          false, a, false, bf, (short)0, acc[w], false, false);
    }
    __syncthreads();
  }

  // C layout: lane holds col n=ln, rows 8h+r.
#pragma unroll
  for (int w = 0; w < 3; ++w)
#pragma unroll
    for (int r = 0; r < 8; ++r)
      qkv[(size_t)w * M * HDD + (size_t)(row + 8 * h + r) * HDD + colB + ln] =
          (__bf16)acc[w][r];
}

// ---------------- Phase 2: flash attention, 1 wave / 16 queries ------------
__global__ __launch_bounds__(32) void flash_attn_kernel(
    const __bf16* __restrict__ qkv, float* __restrict__ out) {
  const int M = BB * TT;
  const __bf16* qb = qkv;
  const __bf16* kb = qkv + (size_t)M * HDD;
  const __bf16* vb = qkv + (size_t)2 * M * HDD;

  __shared__ alignas(16) __bf16 ktile[32 * HDD];  // [key][hd] 4KB
  __shared__ alignas(16) __bf16 vtile[32 * HDD];  // [key][hd] 4KB
  __shared__ alignas(16) __bf16 pT[16 * 32];      // P tile [q][key] 1KB

  const int lane = threadIdx.x;
  const int ln   = lane & 15;
  const int h    = lane >> 4;
  const int b    = blockIdx.y;
  const int qt   = blockIdx.x;          // 0..127
  const int q0   = qt * 16;
  const size_t qrow = (size_t)(b * TT + q0);
  const unsigned ldsK = lds_addr(ktile);
  const unsigned ldsV = lds_addr(vtile);

  // Q fragments (16x64 = two 16x32 A frags), loaded once.
  const v16bf qa0 = load_frag16(qb + (qrow + ln) * HDD + 8 * h);
  const v16bf qa1 = load_frag16(qb + (qrow + ln) * HDD + 32 + 8 * h);

  const v8f zero = {0.f, 0.f, 0.f, 0.f, 0.f, 0.f, 0.f, 0.f};
  v8f oacc[4] = {zero, zero, zero, zero};
  float rmax[8], rsum[8];
#pragma unroll
  for (int r = 0; r < 8; ++r) { rmax[r] = -__builtin_inff(); rsum[r] = 0.f; }
  const float scl = 0.125f;  // 1/sqrt(64)

  for (int k0 = 0; k0 <= q0 + 15; k0 += 32) {
    const size_t krow = (size_t)(b * TT + k0);

    // ---- Async-fill K and V tiles (each a contiguous 4KB bf16 run).
    {
      const uint64_t gK = (uint64_t)(uintptr_t)(kb + krow * HDD);
      const uint64_t gV = (uint64_t)(uintptr_t)(vb + krow * HDD);
#pragma unroll
      for (int j = 0; j < 8; ++j) {
        const unsigned off = (unsigned)(j * 32 + lane) * 16u;
        async_copy_b128(ldsK + off, gK + off);
        async_copy_b128(ldsV + off, gV + off);
      }
      wait_async0();
    }

    // ---- S = Q K^T : two 16x16 key tiles, contraction HD=64 = 2 WMMAs each.
    v8f s0 = zero, s1 = zero;
    {
      v16bf kf = load_frag16(&ktile[ln * HDD + 8 * h]);
      s0 = __builtin_amdgcn_wmma_f32_16x16x32_bf16(false, qa0, false, kf,
                                                   (short)0, s0, false, false);
      kf = load_frag16(&ktile[ln * HDD + 32 + 8 * h]);
      s0 = __builtin_amdgcn_wmma_f32_16x16x32_bf16(false, qa1, false, kf,
                                                   (short)0, s0, false, false);
      kf = load_frag16(&ktile[(16 + ln) * HDD + 8 * h]);
      s1 = __builtin_amdgcn_wmma_f32_16x16x32_bf16(false, qa0, false, kf,
                                                   (short)0, s1, false, false);
      kf = load_frag16(&ktile[(16 + ln) * HDD + 32 + 8 * h]);
      s1 = __builtin_amdgcn_wmma_f32_16x16x32_bf16(false, qa1, false, kf,
                                                   (short)0, s1, false, false);
    }

    // ---- Causal mask + online softmax. C layout: col n=ln, rows 8h+r.
    float mloc[8];
#pragma unroll
    for (int r = 0; r < 8; ++r) {
      const int qr = q0 + 8 * h + r;
      float a0 = (k0 + ln      <= qr) ? s0[r] * scl : -__builtin_inff();
      float a1 = (k0 + 16 + ln <= qr) ? s1[r] * scl : -__builtin_inff();
      s0[r] = a0; s1[r] = a1;
      mloc[r] = fmaxf(a0, a1);
    }
#pragma unroll
    for (int off = 1; off < 16; off <<= 1)
#pragma unroll
      for (int r = 0; r < 8; ++r)
        mloc[r] = fmaxf(mloc[r], __shfl_xor(mloc[r], off, 32));

    float psum[8];
#pragma unroll
    for (int r = 0; r < 8; ++r) {
      const float mnew = fmaxf(rmax[r], mloc[r]);
      const float sc   = __expf(rmax[r] - mnew);  // first chunk: exp(-inf)=0
      rmax[r] = mnew;
      const float p0 = __expf(s0[r] - mnew);      // masked: exp(-inf)=0
      const float p1 = __expf(s1[r] - mnew);
      s0[r] = p0; s1[r] = p1;
      psum[r] = p0 + p1;
      rsum[r] *= sc;
      oacc[0][r] *= sc; oacc[1][r] *= sc; oacc[2][r] *= sc; oacc[3][r] *= sc;
    }
#pragma unroll
    for (int off = 1; off < 16; off <<= 1)
#pragma unroll
      for (int r = 0; r < 8; ++r)
        psum[r] += __shfl_xor(psum[r], off, 32);
#pragma unroll
    for (int r = 0; r < 8; ++r) rsum[r] += psum[r];

    // ---- P: C-layout -> A-layout via LDS round trip (single wave: in-order).
#pragma unroll
    for (int r = 0; r < 8; ++r) {
      pT[(8 * h + r) * 32 + ln]      = (__bf16)s0[r];
      pT[(8 * h + r) * 32 + 16 + ln] = (__bf16)s1[r];
    }
    __syncthreads();  // 1-wave WG: lowers to waitcnt fence (+NOP barrier)
    const v16bf pa = load_frag16(&pT[ln * 32 + 8 * h]);

    // ---- O(16x64) += P(16x32) @ V(32x64): gather V B-frags from row-major
    //      vtile (lane = hd column 16nt+ln, elements = keys 8h+i / 16+8h+i).
#pragma unroll
    for (int nt = 0; nt < 4; ++nt) {
      v16bf vf;
#pragma unroll
      for (int i = 0; i < 8; ++i) {
        vf[i]     = vtile[(8 * h + i) * HDD + 16 * nt + ln];
        vf[8 + i] = vtile[(16 + 8 * h + i) * HDD + 16 * nt + ln];
      }
      oacc[nt] = __builtin_amdgcn_wmma_f32_16x16x32_bf16(
          false, pa, false, vf, (short)0, oacc[nt], false, false);
    }
    __syncthreads();
  }

  // ---- Normalize and store fp32 output [B][T][64].
#pragma unroll
  for (int r = 0; r < 8; ++r) {
    const float inv = 1.0f / rsum[r];
    float* po = out + (qrow + 8 * h + r) * HDD;
    po[ln]      = oacc[0][r] * inv;
    po[16 + ln] = oacc[1][r] * inv;
    po[32 + ln] = oacc[2][r] * inv;
    po[48 + ln] = oacc[3][r] * inv;
  }
}

extern "C" void kernel_launch(void* const* d_in, const int* in_sizes, int n_in,
                              void* d_out, int out_size, void* d_ws, size_t ws_size,
                              hipStream_t stream) {
  const float* x  = (const float*)d_in[0];
  const float* Wq = (const float*)d_in[1];
  const float* Wk = (const float*)d_in[2];
  const float* Wv = (const float*)d_in[3];
  __bf16* qkv = (__bf16*)d_ws;  // 3 * 32768 * 64 bf16 = 12 MB
  float* out = (float*)d_out;

  dim3 g1(BB * TT / 16);               // 2048 tiles, fused q|k|v
  qkv_proj_kernel<<<g1, 128, 0, stream>>>(x, Wq, Wk, Wv, qkv);

  dim3 g2(TT / 16, BB);                // 128 query tiles x 16 batches
  flash_attn_kernel<<<g2, 32, 0, stream>>>(qkv, out);
}
